// ZarvanBlock_55284819034203
// MI455X (gfx1250) — matrix operand order repaired and checked by reading, not compiled, not run on hardware
//
#include <hip/hip_runtime.h>
#include <hip/hip_bf16.h>
#include <math.h>

// ---------------------------------------------------------------------------
// ZarvanBlock for MI455X (gfx1250): bf16 WMMA GEMMs + fused epilogues.
// ---------------------------------------------------------------------------

typedef __bf16 bf16;
typedef __attribute__((ext_vector_type(16))) __bf16 bf16x16;
typedef __attribute__((ext_vector_type(8)))  __bf16 bf16x8;
typedef __attribute__((ext_vector_type(8)))  float  f32x8;

static constexpr int BB  = 4;
static constexpr int SS  = 8192;
static constexpr int EE  = 768;
static constexpr int HIDN = 3072;
static constexpr int TT  = BB * SS;      // 32768 tokens
static constexpr float EPSLN = 1e-5f;
static constexpr float PI_F  = 3.14159265358979323846f;

// epilogue modes
static constexpr int EP_F32       = 0;  // store fp32
static constexpr int EP_GELU_F32  = 1;  // gelu -> fp32
static constexpr int EP_GELU_BF16 = 2;  // gelu -> bf16
static constexpr int EP_SIGMUL    = 3;  // sigmoid(c) * aux -> fp32
static constexpr int EP_RESID     = 4;  // c + aux -> fp32

__device__ __forceinline__ bf16 f2bf(float f) {
    union { float f; unsigned u; } in; in.f = f;
    unsigned r = in.u + 0x7FFFu + ((in.u >> 16) & 1u);   // RNE to bf16
    union { unsigned short s; bf16 b; } out; out.s = (unsigned short)(r >> 16);
    return out.b;
}
__device__ __forceinline__ float bf2f(bf16 b) {
    union { unsigned short s; bf16 b; } in; in.b = b;
    union { unsigned u; float f; } out; out.u = ((unsigned)in.s) << 16;
    return out.f;
}
__device__ __forceinline__ float gelu_exact(float x) {
    return 0.5f * x * (1.0f + erff(x * 0.70710678118654752f));
}
__device__ __forceinline__ float sigm(float x) { return 1.0f / (1.0f + __expf(-x)); }

union FragBF { bf16x16 v; bf16x8 h[2]; };

// ---------------------------------------------------------------------------
// WMMA GEMM: C[M,N] = A[M,K](bf16,row-major) * Bt[N,K](bf16,row-major)^T + bias
// One wave per 64x32 output tile; 4x2 fragment grid; K-step 32.
// A-fragment per-lane K striping: {hi*8..hi*8+7, 16+hi*8..+7} (ISA 7.12.2).
// B-fragment per-lane (column n=lane&15): same K striping as A {hi*8..hi*8+7, 16+hi*8..+7}.
// C-fragment: element e -> M = m0+i*16+hi*8+e, N = n0+j*16+(lane&15).
// ---------------------------------------------------------------------------
template <int EP>
__global__ __launch_bounds__(32)
void wmma_gemm_bf16(const bf16* __restrict__ A, const bf16* __restrict__ Bt,
                    const float* __restrict__ bias,
                    float* __restrict__ outF, bf16* __restrict__ outB,
                    const float* __restrict__ aux,
                    int M, int N, int K)
{
    const int lane = threadIdx.x;
    const int r    = lane & 15;
    const int hi   = lane >> 4;
    const int m0   = blockIdx.y * 64;
    const int n0   = blockIdx.x * 32;

    f32x8 c[4][2];
#pragma unroll
    for (int j = 0; j < 2; ++j) {
        const float bv = bias ? bias[n0 + j * 16 + r] : 0.0f;
#pragma unroll
        for (int i = 0; i < 4; ++i)
#pragma unroll
            for (int e = 0; e < 8; ++e) c[i][j][e] = bv;
    }

    const bf16* arow[4];
    const bf16* brow[2];
#pragma unroll
    for (int i = 0; i < 4; ++i) arow[i] = A  + (size_t)(m0 + i * 16 + r) * K;
#pragma unroll
    for (int j = 0; j < 2; ++j) brow[j] = Bt + (size_t)(n0 + j * 16 + r) * K;

    for (int k = 0; k < K; k += 32) {
        FragBF a[4], b[2];
#pragma unroll
        for (int i = 0; i < 4; ++i) {
            a[i].h[0] = *(const bf16x8*)(arow[i] + k + hi * 8);
            a[i].h[1] = *(const bf16x8*)(arow[i] + k + 16 + hi * 8);
        }
#pragma unroll
        for (int j = 0; j < 2; ++j) {
            b[j].h[0] = *(const bf16x8*)(brow[j] + k + hi * 8);
            b[j].h[1] = *(const bf16x8*)(brow[j] + k + 16 + hi * 8);
        }
        if (k + 64 < K) {   // stream-ahead on the activation rows (global_prefetch)
#pragma unroll
            for (int i = 0; i < 4; ++i) __builtin_prefetch(arow[i] + k + 64, 0, 0);
        }
#pragma unroll
        for (int i = 0; i < 4; ++i)
#pragma unroll
            for (int j = 0; j < 2; ++j)
                c[i][j] = __builtin_amdgcn_wmma_f32_16x16x32_bf16(
                    false, a[i].v, false, b[j].v, (short)0, c[i][j], false, false);
    }

#pragma unroll
    for (int i = 0; i < 4; ++i)
#pragma unroll
        for (int j = 0; j < 2; ++j)
#pragma unroll
            for (int e = 0; e < 8; ++e) {
                const int    m   = m0 + i * 16 + hi * 8 + e;
                const int    n   = n0 + j * 16 + r;
                const size_t idx = (size_t)m * N + n;
                const float  v   = c[i][j][e];
                if      (EP == EP_F32)       outF[idx] = v;
                else if (EP == EP_GELU_F32)  outF[idx] = gelu_exact(v);
                else if (EP == EP_GELU_BF16) outB[idx] = f2bf(gelu_exact(v));
                else if (EP == EP_SIGMUL)    outF[idx] = sigm(v) * aux[idx];
                else if (EP == EP_RESID)     outF[idx] = v + aux[idx];
            }
}

// ---------------------------------------------------------------------------
// Helper kernels
// ---------------------------------------------------------------------------
__global__ void k_f32_to_bf16(const float* __restrict__ in, bf16* __restrict__ out, size_t n) {
    size_t i = (size_t)blockIdx.x * blockDim.x + threadIdx.x;
    if (i < n) out[i] = f2bf(in[i]);
}

// W[K,N] fp32 -> Wt[N,K] bf16
__global__ void k_transpose_bf16(const float* __restrict__ W, bf16* __restrict__ Wt, int K, int N) {
    size_t i = (size_t)blockIdx.x * blockDim.x + threadIdx.x;
    if (i < (size_t)K * N) {
        int n = (int)(i % N), k = (int)(i / N);
        Wt[(size_t)n * K + k] = f2bf(W[i]);
    }
}

// per-row LayerNorm over E=768 -> bf16
__global__ __launch_bounds__(256)
void k_rowln_bf16(const float* __restrict__ in, const float* __restrict__ g,
                  const float* __restrict__ bta, bf16* __restrict__ out) {
    __shared__ float s1[256], s2[256];
    const int row = blockIdx.x, tid = threadIdx.x;
    const float* x = in + (size_t)row * EE;
    float v[3], s = 0.f, ss = 0.f;
#pragma unroll
    for (int c2 = 0; c2 < 3; ++c2) { v[c2] = x[tid + c2 * 256]; s += v[c2]; ss += v[c2] * v[c2]; }
    s1[tid] = s; s2[tid] = ss; __syncthreads();
    for (int off = 128; off > 0; off >>= 1) {
        if (tid < off) { s1[tid] += s1[tid + off]; s2[tid] += s2[tid + off]; }
        __syncthreads();
    }
    const float mean = s1[0] * (1.0f / EE);
    const float var  = s2[0] * (1.0f / EE) - mean * mean;
    const float inv  = rsqrtf(var + EPSLN);
    bf16* o = out + (size_t)row * EE;
#pragma unroll
    for (int c2 = 0; c2 < 3; ++c2) {
        const int e = tid + c2 * 256;
        o[e] = f2bf((v[c2] - mean) * inv * g[e] + bta[e]);
    }
}

// LN then phases = [cos(pi*ns), sin(pi*ns)] -> bf16 (row of 2E)
__global__ __launch_bounds__(256)
void k_ln_trig(const float* __restrict__ in, const float* __restrict__ g,
               const float* __restrict__ bta, bf16* __restrict__ ph) {
    __shared__ float s1[256], s2[256];
    const int row = blockIdx.x, tid = threadIdx.x;
    const float* x = in + (size_t)row * EE;
    float v[3], s = 0.f, ss = 0.f;
#pragma unroll
    for (int c2 = 0; c2 < 3; ++c2) { v[c2] = x[tid + c2 * 256]; s += v[c2]; ss += v[c2] * v[c2]; }
    s1[tid] = s; s2[tid] = ss; __syncthreads();
    for (int off = 128; off > 0; off >>= 1) {
        if (tid < off) { s1[tid] += s1[tid + off]; s2[tid] += s2[tid + off]; }
        __syncthreads();
    }
    const float mean = s1[0] * (1.0f / EE);
    const float var  = s2[0] * (1.0f / EE) - mean * mean;
    const float inv  = rsqrtf(var + EPSLN);
    bf16* o = ph + (size_t)row * (2 * EE);
#pragma unroll
    for (int c2 = 0; c2 < 3; ++c2) {
        const int e = tid + c2 * 256;
        const float ns = (v[c2] - mean) * inv * g[e] + bta[e];
        const float om = ns * PI_F;
        o[e]      = f2bf(cosf(om));
        o[EE + e] = f2bf(sinf(om));
    }
}

// ya[t] = xa[t,:].Was + bas ; g3[t,c] = silu(xa[t,:].Wg3[:,c] + bg3[c]); wave/token
__global__ __launch_bounds__(256)
void k_small_dots(const bf16* __restrict__ xa, const float* __restrict__ Was,
                  const float* __restrict__ bas, const float* __restrict__ Wg3,
                  const float* __restrict__ bg3, float* __restrict__ ya,
                  float* __restrict__ g3o) {
    const int t    = blockIdx.x * 8 + (threadIdx.x >> 5);
    const int lane = threadIdx.x & 31;
    const bf16* xr = xa + (size_t)t * EE;
    float a0 = 0.f, a1 = 0.f, a2 = 0.f, a3 = 0.f;
    for (int k = lane; k < EE; k += 32) {
        const float v = bf2f(xr[k]);
        a0 += v * Was[k];
        a1 += v * Wg3[k * 3 + 0];
        a2 += v * Wg3[k * 3 + 1];
        a3 += v * Wg3[k * 3 + 2];
    }
    for (int off = 16; off > 0; off >>= 1) {
        a0 += __shfl_down(a0, off); a1 += __shfl_down(a1, off);
        a2 += __shfl_down(a2, off); a3 += __shfl_down(a3, off);
    }
    if (lane == 0) {
        ya[t] = a0 + bas[0];
        const float x1 = a1 + bg3[0], x2 = a2 + bg3[1], x3 = a3 + bg3[2];
        g3o[t * 3 + 0] = x1 * sigm(x1);
        g3o[t * 3 + 1] = x2 * sigm(x2);
        g3o[t * 3 + 2] = x3 * sigm(x3);
    }
}

// softmax over S per batch b -> weights w[b,s]
__global__ __launch_bounds__(256)
void k_softmax_seq(const float* __restrict__ ya, float* __restrict__ w) {
    __shared__ float sh[256];
    const int b = blockIdx.x, tid = threadIdx.x;
    const float* y = ya + (size_t)b * SS;
    float m = -3.4e38f;
    for (int s = tid; s < SS; s += 256) m = fmaxf(m, y[s]);
    sh[tid] = m; __syncthreads();
    for (int off = 128; off > 0; off >>= 1) { if (tid < off) sh[tid] = fmaxf(sh[tid], sh[tid + off]); __syncthreads(); }
    m = sh[0]; __syncthreads();
    float l = 0.f;
    for (int s = tid; s < SS; s += 256) l += __expf(y[s] - m);
    sh[tid] = l; __syncthreads();
    for (int off = 128; off > 0; off >>= 1) { if (tid < off) sh[tid] += sh[tid + off]; __syncthreads(); }
    const float inv = 1.0f / sh[0];
    for (int s = tid; s < SS; s += 256) w[(size_t)b * SS + s] = __expf(y[s] - m) * inv;
}

// per (b,e): out[b,e] = sum_s softmax_s(hs[b,s,e]) * hv[b,s,e]  (online)
__global__ __launch_bounds__(256)
void k_colsoftmax_wsum(const float* __restrict__ hs, const float* __restrict__ hv,
                       float* __restrict__ out) {
    __shared__ float sm[256], sl[256], sa[256];
    const int e = blockIdx.x, b = blockIdx.y, tid = threadIdx.x;
    const float* ps = hs + (size_t)b * SS * EE + e;
    const float* pv = hv + (size_t)b * SS * EE + e;
    float m = -3.4e38f, l = 0.f, a = 0.f;
    for (int s = tid; s < SS; s += 256) {
        const float v = ps[(size_t)s * EE];
        const float u = pv[(size_t)s * EE];
        if (v > m) { const float sc = __expf(m - v); l *= sc; a *= sc; m = v; }
        const float ex = __expf(v - m);
        l += ex; a += ex * u;
    }
    sm[tid] = m; sl[tid] = l; sa[tid] = a; __syncthreads();
    for (int off = 128; off > 0; off >>= 1) {
        if (tid < off) {
            const float m2 = sm[tid + off], l2 = sl[tid + off], a2 = sa[tid + off];
            const float M  = fmaxf(sm[tid], m2);
            const float c1 = __expf(sm[tid] - M), c2 = __expf(m2 - M);
            sl[tid] = sl[tid] * c1 + l2 * c2;
            sa[tid] = sa[tid] * c1 + a2 * c2;
            sm[tid] = M;
        }
        __syncthreads();
    }
    if (tid == 0) out[(size_t)b * EE + e] = sa[0] / sl[0];
}

// per (b,e): out[b,e] = sum_s w[b,s] * av[b,s,e]
__global__ __launch_bounds__(256)
void k_col_wsum(const float* __restrict__ w, const float* __restrict__ av,
                float* __restrict__ out) {
    __shared__ float sa[256];
    const int e = blockIdx.x, b = blockIdx.y, tid = threadIdx.x;
    const float* pv = av + (size_t)b * SS * EE + e;
    const float* pw = w + (size_t)b * SS;
    float a = 0.f;
    for (int s = tid; s < SS; s += 256) a += pw[s] * pv[(size_t)s * EE];
    sa[tid] = a; __syncthreads();
    for (int off = 128; off > 0; off >>= 1) { if (tid < off) sa[tid] += sa[tid + off]; __syncthreads(); }
    if (tid == 0) out[(size_t)b * EE + e] = sa[0];
}

// q_h = qraw @ Who + bho   (4 x 768 @ 768 x 768, tiny)
__global__ __launch_bounds__(256)
void k_qh_proj(const float* __restrict__ qraw, const float* __restrict__ Who,
               const float* __restrict__ bho, float* __restrict__ qh) {
    const int idx = blockIdx.x * 256 + threadIdx.x;
    if (idx >= BB * EE) return;
    const int b = idx / EE, n = idx % EE;
    float acc = bho[n];
    const float* qr = qraw + (size_t)b * EE;
    for (int k = 0; k < EE; ++k) acc += qr[k] * Who[(size_t)k * EE + n];
    qh[idx] = acc;
}

// per (b,e): inclusive cumsum over S of gv, /S, write bf16
__global__ __launch_bounds__(256)
void k_scan_div(const float* __restrict__ gv, bf16* __restrict__ accb) {
    __shared__ float sh[256];
    const int e = blockIdx.x, b = blockIdx.y, tid = threadIdx.x;
    const float* p = gv   + (size_t)b * SS * EE + e;
    bf16*        o = accb + (size_t)b * SS * EE + e;
    float carry = 0.f;
    const float invS = 1.0f / (float)SS;
    for (int s0 = 0; s0 < SS; s0 += 256) {
        const float v = p[(size_t)(s0 + tid) * EE];
        sh[tid] = v; __syncthreads();
        for (int off = 1; off < 256; off <<= 1) {
            const float t = (tid >= off) ? sh[tid - off] : 0.f;
            __syncthreads();
            sh[tid] += t; __syncthreads();
        }
        const float incl  = sh[tid];
        const float total = sh[255];
        o[(size_t)(s0 + tid) * EE] = f2bf((carry + incl) * invS);
        carry += total;
        __syncthreads();
    }
}

// h = gh*q_h + ga*q_a + gs*q_s ; LN -> bf16
__global__ __launch_bounds__(256)
void k_combine_ln(const float* __restrict__ qs, const float* __restrict__ qh,
                  const float* __restrict__ qa, const float* __restrict__ g3,
                  const float* __restrict__ g, const float* __restrict__ bta,
                  bf16* __restrict__ out) {
    __shared__ float s1[256], s2[256];
    const int row = blockIdx.x, tid = threadIdx.x;
    const int b   = row >> 13;   // S = 8192
    const float gh = g3[row * 3 + 0], ga = g3[row * 3 + 1], gs = g3[row * 3 + 2];
    const float* q = qs + (size_t)row * EE;
    float h[3], s = 0.f, ss = 0.f;
#pragma unroll
    for (int c2 = 0; c2 < 3; ++c2) {
        const int e = tid + c2 * 256;
        const float v = gh * qh[b * EE + e] + ga * qa[b * EE + e] + gs * q[e];
        h[c2] = v; s += v; ss += v * v;
    }
    s1[tid] = s; s2[tid] = ss; __syncthreads();
    for (int off = 128; off > 0; off >>= 1) {
        if (tid < off) { s1[tid] += s1[tid + off]; s2[tid] += s2[tid + off]; }
        __syncthreads();
    }
    const float mean = s1[0] * (1.0f / EE);
    const float var  = s2[0] * (1.0f / EE) - mean * mean;
    const float inv  = rsqrtf(var + EPSLN);
    bf16* o = out + (size_t)row * EE;
#pragma unroll
    for (int c2 = 0; c2 < 3; ++c2) {
        const int e = tid + c2 * 256;
        o[e] = f2bf((h[c2] - mean) * inv * g[e] + bta[e]);
    }
}

// ---------------------------------------------------------------------------
extern "C" void kernel_launch(void* const* d_in, const int* in_sizes, int n_in,
                              void* d_out, int out_size, void* d_ws, size_t ws_size,
                              hipStream_t stream) {
    (void)in_sizes; (void)n_in; (void)out_size; (void)ws_size;
    const float* x     = (const float*)d_in[0];
    const float* Wa    = (const float*)d_in[1];
    const float* ba    = (const float*)d_in[2];
    const float* lna_g = (const float*)d_in[3];
    const float* lna_b = (const float*)d_in[4];
    const float* Whs   = (const float*)d_in[5];
    const float* bhs   = (const float*)d_in[6];
    const float* Whv   = (const float*)d_in[7];
    const float* bhv   = (const float*)d_in[8];
    const float* Who   = (const float*)d_in[9];
    const float* bho   = (const float*)d_in[10];
    const float* Was   = (const float*)d_in[11];
    const float* bas   = (const float*)d_in[12];
    const float* Wav   = (const float*)d_in[13];
    const float* bav   = (const float*)d_in[14];
    const float* Wg    = (const float*)d_in[15];
    const float* bg    = (const float*)d_in[16];
    const float* Wv    = (const float*)d_in[17];
    const float* bv    = (const float*)d_in[18];
    const float* Wp    = (const float*)d_in[19];
    const float* bp    = (const float*)d_in[20];
    const float* lns_g = (const float*)d_in[21];
    const float* lns_b = (const float*)d_in[22];
    const float* Wo2   = (const float*)d_in[23];
    const float* bo2   = (const float*)d_in[24];
    const float* Wg3   = (const float*)d_in[25];
    const float* bg3   = (const float*)d_in[26];
    const float* W1    = (const float*)d_in[27];
    const float* b1    = (const float*)d_in[28];
    const float* W2    = (const float*)d_in[29];
    const float* b2    = (const float*)d_in[30];
    const float* lno_g = (const float*)d_in[31];
    const float* lno_b = (const float*)d_in[32];
    float* out = (float*)d_out;

    // ---- workspace arena -------------------------------------------------
    char* ws = (char*)d_ws;
    size_t off = 0;
    auto alloc = [&](size_t bytes) -> char* {
        char* p = ws + off;
        off += (bytes + 255) & ~(size_t)255;
        return p;
    };
    bf16* wtWa  = (bf16*)alloc((size_t)EE * EE * 2);
    bf16* wtWhs = (bf16*)alloc((size_t)EE * EE * 2);
    bf16* wtWhv = (bf16*)alloc((size_t)EE * EE * 2);
    bf16* wtWav = (bf16*)alloc((size_t)EE * EE * 2);
    bf16* wtWg  = (bf16*)alloc((size_t)EE * EE * 2);
    bf16* wtWv  = (bf16*)alloc((size_t)EE * EE * 2);
    bf16* wtWp  = (bf16*)alloc((size_t)EE * EE * 2);
    bf16* wtWo2 = (bf16*)alloc((size_t)EE * (2 * EE) * 2);     // 768 x 1536
    bf16* wtW1  = (bf16*)alloc((size_t)HIDN * EE * 2);         // 3072 x 768
    bf16* wtW2  = (bf16*)alloc((size_t)EE * HIDN * 2);         // 768 x 3072
    bf16* xbf   = (bf16*)alloc((size_t)TT * EE * 2);           // x bf16; reused as acc bf16
    bf16* xabf  = (bf16*)alloc((size_t)TT * EE * 2);           // xa bf16; reused as hn bf16
    float* f1   = (float*)alloc((size_t)TT * EE * 4);          // gelu(x@Wa) -> yp
    float* f2   = (float*)alloc((size_t)TT * EE * 4);          // hs -> av -> gv -> q_s
    float* f3   = (float*)alloc((size_t)TT * EE * 4);          // hv -> vals
    bf16* phbf  = (bf16*)alloc((size_t)TT * 2 * EE * 2);       // phases
    bf16* ff1bf = (bf16*)alloc((size_t)TT * HIDN * 2);         // gelu FFN hidden
    float* bya  = (float*)alloc((size_t)TT * 4);
    float* bw   = (float*)alloc((size_t)TT * 4);
    float* bg3b = (float*)alloc((size_t)TT * 3 * 4);
    float* bqraw= (float*)alloc((size_t)BB * EE * 4);
    float* bqh  = (float*)alloc((size_t)BB * EE * 4);
    float* bqa  = (float*)alloc((size_t)BB * EE * 4);

    // ---- one-time conversions -------------------------------------------
    {
        size_t n = (size_t)TT * EE;
        k_f32_to_bf16<<<(unsigned)((n + 255) / 256), 256, 0, stream>>>(x, xbf, n);
    }
    auto tw = [&](const float* W, bf16* Wt, int K, int N) {
        size_t n = (size_t)K * N;
        k_transpose_bf16<<<(unsigned)((n + 255) / 256), 256, 0, stream>>>(W, Wt, K, N);
    };
    tw(Wa,  wtWa,  EE, EE);   tw(Whs, wtWhs, EE, EE);   tw(Whv, wtWhv, EE, EE);
    tw(Wav, wtWav, EE, EE);   tw(Wg,  wtWg,  EE, EE);   tw(Wv,  wtWv,  EE, EE);
    tw(Wp,  wtWp,  EE, EE);   tw(Wo2, wtWo2, 2 * EE, EE);
    tw(W1,  wtW1,  EE, HIDN); tw(W2,  wtW2,  HIDN, EE);

    // ---- GEMM launcher ---------------------------------------------------
    dim3 blk(32);
    auto gemm = [&](int ep, const bf16* A, const bf16* Bt, const float* bias,
                    float* oF, bf16* oB, const float* aux, int M, int N, int K) {
        dim3 grid(N / 32, M / 64);
        switch (ep) {
        case EP_F32:       wmma_gemm_bf16<EP_F32>      <<<grid, blk, 0, stream>>>(A, Bt, bias, oF, oB, aux, M, N, K); break;
        case EP_GELU_F32:  wmma_gemm_bf16<EP_GELU_F32> <<<grid, blk, 0, stream>>>(A, Bt, bias, oF, oB, aux, M, N, K); break;
        case EP_GELU_BF16: wmma_gemm_bf16<EP_GELU_BF16><<<grid, blk, 0, stream>>>(A, Bt, bias, oF, oB, aux, M, N, K); break;
        case EP_SIGMUL:    wmma_gemm_bf16<EP_SIGMUL>   <<<grid, blk, 0, stream>>>(A, Bt, bias, oF, oB, aux, M, N, K); break;
        case EP_RESID:     wmma_gemm_bf16<EP_RESID>    <<<grid, blk, 0, stream>>>(A, Bt, bias, oF, oB, aux, M, N, K); break;
        }
    };

    // 1) xa = LN(gelu(x @ Wa + ba))
    gemm(EP_GELU_F32, xbf, wtWa, ba, f1, nullptr, nullptr, TT, EE, EE);
    k_rowln_bf16<<<TT, 256, 0, stream>>>(f1, lna_g, lna_b, xabf);

    // 2) small dots: ya (for asc softmax) and g3 = silu(xa@Wg3+bg3)
    k_small_dots<<<TT / 8, 256, 0, stream>>>(xabf, Was, bas, Wg3, bg3, bya, bg3b);
    k_softmax_seq<<<BB, 256, 0, stream>>>(bya, bw);

    // 3) q_h branch: hs, hv, column softmax-weighted sum, then @Who
    gemm(EP_F32, xabf, wtWhs, bhs, f2, nullptr, nullptr, TT, EE, EE);
    gemm(EP_F32, xabf, wtWhv, bhv, f3, nullptr, nullptr, TT, EE, EE);
    k_colsoftmax_wsum<<<dim3(EE, BB), 256, 0, stream>>>(f2, f3, bqraw);
    k_qh_proj<<<(BB * EE) / 256, 256, 0, stream>>>(bqraw, Who, bho, bqh);

    // 4) q_a branch: av then weighted sum by asc
    gemm(EP_F32, xabf, wtWav, bav, f2, nullptr, nullptr, TT, EE, EE);
    k_col_wsum<<<dim3(EE, BB), 256, 0, stream>>>(bw, f2, bqa);

    // 5) spectral branch: gv = sigmoid(xa@Wg+bg) * (xa@Wv+bv); cumsum/S
    gemm(EP_F32, xabf, wtWv, bv, f3, nullptr, nullptr, TT, EE, EE);
    gemm(EP_SIGMUL, xabf, wtWg, bg, f2, nullptr, f3, TT, EE, EE);
    k_scan_div<<<dim3(EE, BB), 256, 0, stream>>>(f2, xbf);     // xbf reused as acc bf16
    gemm(EP_F32, xbf, wtWp, bp, f1, nullptr, nullptr, TT, EE, EE);
    k_ln_trig<<<TT, 256, 0, stream>>>(f1, lns_g, lns_b, phbf);
    gemm(EP_F32, phbf, wtWo2, bo2, f2, nullptr, nullptr, TT, EE, 2 * EE);  // q_s

    // 6) combine gates + LN -> hn (bf16, reuse xabf)
    k_combine_ln<<<TT, 256, 0, stream>>>(f2, bqh, bqa, bg3b, lno_g, lno_b, xabf);

    // 7) FFN + residual
    gemm(EP_GELU_BF16, xabf, wtW1, b1, nullptr, ff1bf, nullptr, TT, HIDN, EE);
    gemm(EP_RESID, ff1bf, wtW2, b2, out, nullptr, x, TT, EE, HIDN);
}
